// ALMCoder_23270132810292
// MI455X (gfx1250) — matrix operand order repaired and checked by reading, not compiled, not run on hardware
//
#include <hip/hip_runtime.h>

// Problem constants (from reference)
#define Tt   1024
#define Cc   1024
#define Hh   16
#define KVh  8
#define Dd   64
#define Ff   2048
#define Aa   10
#define NEx  20      // A * EPA
#define TK   3
#define Ll   2
#define Vv   32000

typedef __attribute__((ext_vector_type(16))) _Float16 v16h;
typedef __attribute__((ext_vector_type(8)))  _Float16 v8h;
typedef __attribute__((ext_vector_type(8)))  float    v8f;

// Async-DMA pointer types: builtin expects int4* in AS1 (global) / AS3 (LDS).
typedef int v4i __attribute__((vector_size(16)));
typedef __attribute__((address_space(1))) v4i* p1v4i;
typedef __attribute__((address_space(3))) v4i* p3v4i;

// CDNA5 async global->LDS (ASYNCcnt) if this toolchain exposes the builtins.
#if defined(__has_builtin)
#if __has_builtin(__builtin_amdgcn_global_load_async_to_lds_b128) && \
    __has_builtin(__builtin_amdgcn_s_wait_asynccnt)
#define USE_ASYNC_LDS 1
#endif
#endif
#ifndef USE_ASYNC_LDS
#define USE_ASYNC_LDS 0
#endif

__device__ __forceinline__ v8f wmma16(v16h a, v16h b, v8f c) {
  return __builtin_amdgcn_wmma_f32_16x16x32_f16(false, a, false, b, (short)0, c,
                                                false, false);
}

__device__ __forceinline__ float sigf(float x) { return 1.0f / (1.0f + __expf(-x)); }

__device__ __forceinline__ v8h ld8(const _Float16* p) { return *(const v8h*)p; }
__device__ __forceinline__ void st8(_Float16* p, v8h v) { *(v8h*)p = v; }

#if USE_ASYNC_LDS
__device__ __forceinline__ void async_cp16B(const _Float16* gsrc, _Float16* ldst) {
  __builtin_amdgcn_global_load_async_to_lds_b128((p1v4i)(void*)gsrc,
                                                 (p3v4i)(void*)ldst, 0, 0);
}
#endif

// ---------------------------------------------------------------------------
// WMMA GEMM: C[M][N](f32) = A[M][K](f16, row-major) * Bt[N][K](f16, B transposed)
// BM=128 BN=128 BK=32, 256 threads = 8 waves (4 in M x 2 in N), 32x64 per wave
// (2x4 WMMA tiles). Double-buffered LDS; async global->LDS DMA when available,
// otherwise register-staged software pipeline. One barrier per k-step.
// MODE: 0=store 1=tanh 2=sigmoid 3=accumulate 4=accumulate+sigmoid
// ---------------------------------------------------------------------------
#define BM 128
#define BN 128
#define BK 32

template <int MODE>
__global__ __launch_bounds__(256) void k_gemm(const _Float16* __restrict__ A,
                                              const _Float16* __restrict__ Bt,
                                              float* __restrict__ Cout,
                                              int M, int N, int K) {
  __shared__ __attribute__((aligned(16))) _Float16 As[2][BM * BK];
  __shared__ __attribute__((aligned(16))) _Float16 Bs[2][BN * BK];

  const int tid  = threadIdx.x;
  const int lane = tid & 31;
  const int wave = tid >> 5;
  const int wm   = wave & 3;   // 4 waves in M
  const int wn   = wave >> 2;  // 2 waves in N
  const int hi   = lane >> 4;  // 0/1
  const int lr   = lane & 15;
  const int tileM = blockIdx.y * BM;
  const int tileN = blockIdx.x * BN;

  // Staging map: 256 threads, each moves 16 halves of A row r and 16 of B row r.
  const int r  = tid >> 1;          // 0..127
  const int sg = (tid & 1) * 16;    // 0,16

  const _Float16* gA = A  + (size_t)(tileM + r) * K + sg;
  const _Float16* gB = Bt + (size_t)(tileN + r) * K + sg;
  _Float16* lA = &As[0][r * BK + sg];
  _Float16* lB = &Bs[0][r * BK + sg];
  const int lbuf = BM * BK;  // halves per buffer (As and Bs identical size)

  v8f acc[2][4] = {};

#if USE_ASYNC_LDS
  // ---- async-DMA double-buffered pipeline ----
  async_cp16B(gA,     lA);
  async_cp16B(gA + 8, lA + 8);
  async_cp16B(gB,     lB);
  async_cp16B(gB + 8, lB + 8);
  __builtin_amdgcn_s_wait_asynccnt(0);
  __syncthreads();
#else
  // ---- register-staged prologue ----
  {
    v8h a0 = ld8(gA), a1 = ld8(gA + 8);
    v8h b0 = ld8(gB), b1 = ld8(gB + 8);
    st8(lA, a0); st8(lA + 8, a1);
    st8(lB, b0); st8(lB + 8, b1);
  }
  __syncthreads();
#endif

  int buf = 0;
  for (int k0 = 0; k0 < K; k0 += BK) {
    const int  nxt  = buf ^ 1;
    const bool more = (k0 + BK) < K;

#if USE_ASYNC_LDS
    if (more) {
      const _Float16* nA = gA + (k0 + BK);
      const _Float16* nB = gB + (k0 + BK);
      async_cp16B(nA,     lA + nxt * lbuf);
      async_cp16B(nA + 8, lA + nxt * lbuf + 8);
      async_cp16B(nB,     lB + nxt * lbuf);
      async_cp16B(nB + 8, lB + nxt * lbuf + 8);
    }
#else
    v8h na0, na1, nb0, nb1;
    if (more) {
      na0 = ld8(gA + (k0 + BK));
      na1 = ld8(gA + (k0 + BK) + 8);
      nb0 = ld8(gB + (k0 + BK));
      nb1 = ld8(gB + (k0 + BK) + 8);
    }
    if (k0 + 2 * BK < K) {
      __builtin_prefetch(gA + (k0 + 2 * BK), 0, 1);
      __builtin_prefetch(gB + (k0 + 2 * BK), 0, 1);
    }
#endif

    // ---- compute current buffer ----
    // A fragment (16-bit A 16x32): lanes0-15 -> K0..7 & K16..23,
    // lanes16-31 -> K8..15 & K24..31.
    v16h af[2], bf[4];
#pragma unroll
    for (int mi = 0; mi < 2; ++mi) {
      int row = wm * 32 + mi * 16 + lr;
      v8h lo = *(const v8h*)(&As[buf][row * BK + hi * 8]);
      v8h h8 = *(const v8h*)(&As[buf][row * BK + hi * 8 + 16]);
      af[mi] = __builtin_shufflevector(lo, h8, 0, 1, 2, 3, 4, 5, 6, 7,
                                       8, 9, 10, 11, 12, 13, 14, 15);
    }
    // B fragment (32x16): lanes0-15 -> K0..7 & K16..23, lanes16-31 -> K8..15 & K24..31.
#pragma unroll
    for (int ni = 0; ni < 4; ++ni) {
      int col = wn * 64 + ni * 16 + lr;
      v8h lo = *(const v8h*)(&Bs[buf][col * BK + hi * 8]);
      v8h h8 = *(const v8h*)(&Bs[buf][col * BK + hi * 8 + 16]);
      bf[ni] = __builtin_shufflevector(lo, h8, 0, 1, 2, 3, 4, 5, 6, 7,
                                       8, 9, 10, 11, 12, 13, 14, 15);
    }
#pragma unroll
    for (int mi = 0; mi < 2; ++mi)
#pragma unroll
      for (int ni = 0; ni < 4; ++ni)
        acc[mi][ni] = wmma16(af[mi], bf[ni], acc[mi][ni]);

#if USE_ASYNC_LDS
    if (more) __builtin_amdgcn_s_wait_asynccnt(0);
#else
    if (more) {
      st8(lA + nxt * lbuf, na0); st8(lA + nxt * lbuf + 8, na1);
      st8(lB + nxt * lbuf, nb0); st8(lB + nxt * lbuf + 8, nb1);
    }
#endif
    __syncthreads();
    buf = nxt;
  }

  // Store: C/D layout -> VGPR rr: lanes0-15 M=rr, lanes16-31 M=rr+8; N=lane%16
#pragma unroll
  for (int mi = 0; mi < 2; ++mi)
#pragma unroll
    for (int ni = 0; ni < 4; ++ni) {
      int rb = tileM + wm * 32 + mi * 16 + hi * 8;
      int cb = tileN + wn * 64 + ni * 16 + lr;
#pragma unroll
      for (int rr = 0; rr < 8; ++rr) {
        size_t o = (size_t)(rb + rr) * N + cb;
        float v = acc[mi][ni][rr];
        if (MODE == 0) Cout[o] = v;
        else if (MODE == 1) Cout[o] = tanhf(v);
        else if (MODE == 2) Cout[o] = sigf(v);
        else if (MODE == 3) Cout[o] += v;
        else if (MODE == 4) Cout[o] = sigf(Cout[o] + v);
      }
    }
}

// ---------------------------------------------------------------------------
// Elementwise / small kernels
// ---------------------------------------------------------------------------
__global__ void k_transpose(const float* __restrict__ in, _Float16* __restrict__ out,
                            int K, int N) {
  size_t i = (size_t)blockIdx.x * blockDim.x + threadIdx.x;
  size_t n = (size_t)K * N;
  if (i >= n) return;
  int k = (int)(i / N), c = (int)(i % N);
  out[(size_t)c * K + k] = (_Float16)in[i];
}

__global__ void k_convert(const float* __restrict__ in, _Float16* __restrict__ out,
                          size_t n) {
  size_t i = (size_t)blockIdx.x * blockDim.x + threadIdx.x;
  if (i < n) out[i] = (_Float16)in[i];
}

__global__ void k_gather(const int* __restrict__ idx, const float* __restrict__ wte,
                         float* __restrict__ x) {
  size_t i = (size_t)blockIdx.x * blockDim.x + threadIdx.x;
  if (i >= (size_t)Tt * Cc) return;
  int t = (int)(i / Cc), c = (int)(i % Cc);
  x[i] = wte[(size_t)idx[t] * Cc + c];
}

__global__ void k_add(float* __restrict__ x, const float* __restrict__ y, size_t n) {
  size_t i = (size_t)blockIdx.x * blockDim.x + threadIdx.x;
  if (i < n) x[i] += y[i];
}

__global__ __launch_bounds__(256) void k_ln(const float* __restrict__ x,
                                            const float* __restrict__ g,
                                            const float* __restrict__ b,
                                            float* __restrict__ out,
                                            _Float16* __restrict__ out16) {
  int t = blockIdx.x;
  const float* row = x + (size_t)t * Cc;
  __shared__ float s1[8], s2[8];
  float sum = 0.f, sq = 0.f;
  for (int c = threadIdx.x; c < Cc; c += 256) { float v = row[c]; sum += v; sq += v * v; }
  for (int off = 16; off > 0; off >>= 1) {
    sum += __shfl_down(sum, off, 32);
    sq  += __shfl_down(sq,  off, 32);
  }
  if ((threadIdx.x & 31) == 0) { s1[threadIdx.x >> 5] = sum; s2[threadIdx.x >> 5] = sq; }
  __syncthreads();
  float ts = 0.f, tq = 0.f;
  for (int i = 0; i < 8; ++i) { ts += s1[i]; tq += s2[i]; }
  float mu  = ts / Cc;
  float var = tq / Cc - mu * mu;
  float inv = rsqrtf(var + 1e-5f);
  for (int c = threadIdx.x; c < Cc; c += 256) {
    float v = (row[c] - mu) * inv * g[c] + b[c];
    out[(size_t)t * Cc + c]   = v;
    out16[(size_t)t * Cc + c] = (_Float16)v;
  }
}

__global__ void k_rope(float* __restrict__ buf, int NH) {
  size_t i = (size_t)blockIdx.x * blockDim.x + threadIdx.x;
  size_t n = (size_t)Tt * NH * 32;
  if (i >= n) return;
  int f = (int)(i % 32);
  int h = (int)((i / 32) % NH);
  int t = (int)(i / (32 * NH));
  float* p = buf + ((size_t)t * NH + h) * 64;
  float inv = __powf(10000.f, -(2.f * f) / 64.f);
  float ang = (float)t * inv;
  float cs = __cosf(ang), sn = __sinf(ang);
  float x1 = p[f], x2 = p[f + 32];
  p[f]      = x1 * cs - x2 * sn;
  p[f + 32] = x2 * cs + x1 * sn;
}

__global__ __launch_bounds__(128) void k_attn(const float* __restrict__ q,
                                              const float* __restrict__ kbuf,
                                              const float* __restrict__ vbuf,
                                              float* __restrict__ y) {
  int tq = blockIdx.x, h = blockIdx.y;
  int kvh = h >> 1;  // H/KV = 2
  __shared__ float qs[64];
  __shared__ float sc[Tt];
  __shared__ float red[4];
  int tid = threadIdx.x;
  if (tid < 64) qs[tid] = q[((size_t)tq * Hh + h) * 64 + tid];
  __syncthreads();
  float lmax = -1e30f;
  for (int s = tid; s <= tq; s += 128) {
    const float* kr = kbuf + ((size_t)s * KVh + kvh) * 64;
    float d = 0.f;
#pragma unroll
    for (int i = 0; i < 64; ++i) d += qs[i] * kr[i];
    d *= 0.125f;
    sc[s] = d;
    lmax = fmaxf(lmax, d);
  }
  for (int off = 16; off > 0; off >>= 1) lmax = fmaxf(lmax, __shfl_xor(lmax, off, 32));
  if ((tid & 31) == 0) red[tid >> 5] = lmax;
  __syncthreads();
  float mx = fmaxf(fmaxf(red[0], red[1]), fmaxf(red[2], red[3]));
  float lsum = 0.f;
  for (int s = tid; s <= tq; s += 128) { float e = __expf(sc[s] - mx); sc[s] = e; lsum += e; }
  for (int off = 16; off > 0; off >>= 1) lsum += __shfl_xor(lsum, off, 32);
  __syncthreads();
  if ((tid & 31) == 0) red[tid >> 5] = lsum;
  __syncthreads();
  float inv = 1.f / (red[0] + red[1] + red[2] + red[3]);
  if (tid < 64) {
    float acc = 0.f;
    for (int s = 0; s <= tq; ++s) acc += sc[s] * vbuf[((size_t)s * KVh + kvh) * 64 + tid];
    y[((size_t)tq * Hh + h) * 64 + tid] = acc * inv;
  }
}

__global__ __launch_bounds__(256) void k_gate(const float* __restrict__ h2,
                                              const float* __restrict__ ag,
                                              const float* __restrict__ eg,
                                              float* __restrict__ exp_w,
                                              float* __restrict__ topw,
                                              int* __restrict__ topi) {
  int t = blockIdx.x;
  __shared__ float dots[32];
  if (threadIdx.x < 32) dots[threadIdx.x] = 0.f;
  __syncthreads();
  const float* row = h2 + (size_t)t * Cc;
  float loc[30];
#pragma unroll
  for (int j = 0; j < 30; ++j) loc[j] = 0.f;
  for (int c = threadIdx.x; c < Cc; c += 256) {
    float v = row[c];
#pragma unroll
    for (int a = 0; a < Aa; ++a) loc[a] += v * ag[(size_t)c * Aa + a];
#pragma unroll
    for (int e = 0; e < NEx; ++e) loc[Aa + e] += v * eg[(size_t)c * NEx + e];
  }
#pragma unroll
  for (int j = 0; j < 30; ++j) atomicAdd(&dots[j], loc[j]);
  __syncthreads();
  if (threadIdx.x == 0) {
    float mx = -1e30f;
    for (int a = 0; a < Aa; ++a) mx = fmaxf(mx, dots[a]);
    float p[Aa], sm = 0.f;
    for (int a = 0; a < Aa; ++a) { p[a] = __expf(dots[a] - mx); sm += p[a]; }
    for (int a = 0; a < Aa; ++a) p[a] /= sm;
    int ord[Aa];
    for (int a = 0; a < Aa; ++a) ord[a] = a;
    float wsum = 0.f;
    for (int k = 0; k < TK; ++k) {
      int best = k;
      for (int a = k + 1; a < Aa; ++a) if (p[ord[a]] > p[ord[best]]) best = a;
      int tmp = ord[k]; ord[k] = ord[best]; ord[best] = tmp;
      topi[t * TK + k] = ord[k];
      topw[t * TK + k] = p[ord[k]];
      wsum += p[ord[k]];
    }
    for (int k = 0; k < TK; ++k) topw[t * TK + k] /= wsum;
    for (int a = 0; a < Aa; ++a) {
      float x0 = dots[Aa + a * 2], x1 = dots[Aa + a * 2 + 1];
      float m = fmaxf(x0, x1);
      float e0 = __expf(x0 - m), e1 = __expf(x1 - m), s = e0 + e1;
      exp_w[t * NEx + a * 2]     = e0 / s;
      exp_w[t * NEx + a * 2 + 1] = e1 / s;
    }
  }
}

__global__ void k_silu_mul(const float* __restrict__ u, const float* __restrict__ g,
                           _Float16* __restrict__ act, size_t n) {
  size_t i = (size_t)blockIdx.x * blockDim.x + threadIdx.x;
  if (i >= n) return;
  float uv = u[i];
  act[i] = (_Float16)((uv * sigf(uv)) * g[i]);
}

__global__ void k_init_agent(float* __restrict__ agent_out, const float* __restrict__ role) {
  size_t i = (size_t)blockIdx.x * blockDim.x + threadIdx.x;
  size_t n = (size_t)Tt * Aa * Cc;
  if (i >= n) return;
  int a = (int)((i / Cc) % Aa);
  agent_out[i] = role[(size_t)a * Cc + (i % Cc)];
}

__global__ void k_acc_agent(const float* __restrict__ ye, const float* __restrict__ exp_w,
                            float* __restrict__ agent_out, int e, int a) {
  size_t i = (size_t)blockIdx.x * blockDim.x + threadIdx.x;
  if (i >= (size_t)Tt * Cc) return;
  int t = (int)(i / Cc);
  agent_out[((size_t)t * Aa + a) * Cc + (i % Cc)] += exp_w[t * NEx + e] * ye[i];
}

__global__ void k_build_wout(const float* __restrict__ agent_out,
                             const float* __restrict__ topw, const int* __restrict__ topi,
                             float* __restrict__ w_out, _Float16* __restrict__ w_out16) {
  size_t i = (size_t)blockIdx.x * blockDim.x + threadIdx.x;
  size_t n = (size_t)Tt * TK * Cc;
  if (i >= n) return;
  int t = (int)(i / (TK * Cc));
  int rr = (int)(i % (TK * Cc));
  int k = rr / Cc, c = rr % Cc;
  int a = topi[t * TK + k];
  float v = topw[t * TK + k] * agent_out[((size_t)t * Aa + a) * Cc + c];
  w_out[i]   = v;
  w_out16[i] = (_Float16)v;
}

__global__ void k_other(const float* __restrict__ msgs, float* __restrict__ other,
                        _Float16* __restrict__ other16) {
  size_t i = (size_t)blockIdx.x * blockDim.x + threadIdx.x;
  size_t n = (size_t)Tt * TK * Cc;
  if (i >= n) return;
  int t = (int)(i / (TK * Cc));
  int c = (int)(i % Cc);
  size_t b0 = ((size_t)t * TK) * Cc + c;
  float s = msgs[b0] + msgs[b0 + Cc] + msgs[b0 + 2 * Cc];
  float v = s - msgs[i];
  other[i]   = v;
  other16[i] = (_Float16)v;
}

__global__ void k_strength(const int* __restrict__ topi, const float* __restrict__ ts,
                           float* __restrict__ strength) {
  if (threadIdx.x < TK) {
    int sid = topi[(Tt - 1) * TK + threadIdx.x];
    strength[threadIdx.x] = sigf(ts[sid]);
  }
}

__global__ void k_refine(float* __restrict__ x, const float* __restrict__ w_out,
                         const float* __restrict__ gate, const float* __restrict__ lst,
                         const float* __restrict__ strength) {
  size_t i = (size_t)blockIdx.x * blockDim.x + threadIdx.x;
  if (i >= (size_t)Tt * Cc) return;
  int t = (int)(i / Cc), c = (int)(i % Cc);
  float acc = 0.f;
#pragma unroll
  for (int k = 0; k < TK; ++k) {
    size_t b = ((size_t)t * TK + k) * Cc + c;
    acc += w_out[b] + strength[k] * gate[b] * lst[b];
  }
  x[i] += acc;
}

// ---------------------------------------------------------------------------
// Host orchestration
// ---------------------------------------------------------------------------
static inline unsigned gblk(size_t n) { return (unsigned)((n + 255) / 256); }

extern "C" void kernel_launch(void* const* d_in, const int* in_sizes, int n_in,
                              void* d_out, int out_size, void* d_ws, size_t ws_size,
                              hipStream_t stream) {
  (void)in_sizes; (void)n_in; (void)out_size; (void)ws_size;
  const int*   idx  = (const int*)  d_in[0];
  const float* wte  = (const float*)d_in[1];
  const float* ln1g = (const float*)d_in[2];
  const float* ln1b = (const float*)d_in[3];
  const float* wq   = (const float*)d_in[4];
  const float* wk   = (const float*)d_in[5];
  const float* wv   = (const float*)d_in[6];
  const float* wo   = (const float*)d_in[7];
  const float* ln2g = (const float*)d_in[8];
  const float* ln2b = (const float*)d_in[9];
  const float* agate= (const float*)d_in[10];
  const float* egate= (const float*)d_in[11];
  const float* role = (const float*)d_in[12];
  const float* ew1  = (const float*)d_in[13];
  const float* ew2  = (const float*)d_in[14];
  const float* ew3  = (const float*)d_in[15];
  const float* msgw = (const float*)d_in[16];
  const float* lisw = (const float*)d_in[17];
  const float* tgw  = (const float*)d_in[18];
  const float* tstr = (const float*)d_in[19];
  const float* lnfg = (const float*)d_in[20];
  const float* lnfb = (const float*)d_in[21];
  const float* lmh  = (const float*)d_in[22];
  float* out = (float*)d_out;

  const int HD = Hh * Dd;   // 1024
  const int KD = KVh * Dd;  // 512
  const int T3 = Tt * TK;   // 3072

  char* ws = (char*)d_ws;
  size_t off = 0;
  auto alloc = [&](size_t bytes) -> void* {
    void* p = ws + off;
    off += (bytes + 255) & ~(size_t)255;
    return p;
  };

  // f16 transposed weights [N][K]
  _Float16* wqT  = (_Float16*)alloc((size_t)Ll * HD * Cc * 2);
  _Float16* wkT  = (_Float16*)alloc((size_t)Ll * KD * Cc * 2);
  _Float16* wvT  = (_Float16*)alloc((size_t)Ll * KD * Cc * 2);
  _Float16* woT  = (_Float16*)alloc((size_t)Ll * Cc * HD * 2);
  _Float16* w1T  = (_Float16*)alloc((size_t)Ll * NEx * Ff * Cc * 2);
  _Float16* w3T  = (_Float16*)alloc((size_t)Ll * NEx * Ff * Cc * 2);
  _Float16* w2T  = (_Float16*)alloc((size_t)Ll * NEx * Cc * Ff * 2);
  _Float16* msgT = (_Float16*)alloc((size_t)Ll * Cc * Cc * 2);
  _Float16* lisT = (_Float16*)alloc((size_t)Ll * Cc * Cc * 2);
  _Float16* tgTopT = (_Float16*)alloc((size_t)Ll * Cc * Cc * 2);
  _Float16* tgBotT = (_Float16*)alloc((size_t)Ll * Cc * Cc * 2);
  _Float16* lmT  = (_Float16*)alloc((size_t)Vv * Cc * 2);

  // activations
  float*    x      = (float*)alloc((size_t)Tt * Cc * 4);
  float*    h      = (float*)alloc((size_t)Tt * Cc * 4);
  _Float16* h16    = (_Float16*)alloc((size_t)Tt * Cc * 2);
  float*    h2     = (float*)alloc((size_t)Tt * Cc * 4);
  _Float16* h2_16  = (_Float16*)alloc((size_t)Tt * Cc * 2);
  float*    qb     = (float*)alloc((size_t)Tt * HD * 4);
  float*    kb     = (float*)alloc((size_t)Tt * KD * 4);
  float*    vb     = (float*)alloc((size_t)Tt * KD * 4);
  float*    yb     = (float*)alloc((size_t)Tt * HD * 4);
  _Float16* y16    = (_Float16*)alloc((size_t)Tt * HD * 2);
  float*    ao     = (float*)alloc((size_t)Tt * Cc * 4);
  float*    ub     = (float*)alloc((size_t)Tt * Ff * 4);
  float*    gb     = (float*)alloc((size_t)Tt * Ff * 4);
  _Float16* act16  = (_Float16*)alloc((size_t)Tt * Ff * 2);
  float*    yeb    = (float*)alloc((size_t)Tt * Cc * 4);
  float*    expw   = (float*)alloc((size_t)Tt * NEx * 4);
  float*    topw   = (float*)alloc((size_t)Tt * TK * 4);
  int*      topi   = (int*)  alloc((size_t)Tt * TK * 4);
  float*    agout  = (float*)alloc((size_t)Tt * Aa * Cc * 4);
  float*    wout   = (float*)alloc((size_t)T3 * Cc * 4);
  _Float16* wout16 = (_Float16*)alloc((size_t)T3 * Cc * 2);
  float*    msgs   = (float*)alloc((size_t)T3 * Cc * 4);
  float*    othr   = (float*)alloc((size_t)T3 * Cc * 4);
  _Float16* othr16 = (_Float16*)alloc((size_t)T3 * Cc * 2);
  float*    lst    = (float*)alloc((size_t)T3 * Cc * 4);
  _Float16* lst16  = (_Float16*)alloc((size_t)T3 * Cc * 2);
  float*    gateb  = (float*)alloc((size_t)T3 * Cc * 4);
  float*    strg   = (float*)alloc(256);

  auto xpose = [&](const float* in, _Float16* o, int K, int N) {
    size_t n = (size_t)K * N;
    k_transpose<<<gblk(n), 256, 0, stream>>>(in, o, K, N);
  };
  auto gemm = [&](int mode, const _Float16* A, const _Float16* Bt, float* C_,
                  int M, int N, int K) {
    dim3 g((unsigned)(N / BN), (unsigned)(M / BM));
    switch (mode) {
      case 0: k_gemm<0><<<g, 256, 0, stream>>>(A, Bt, C_, M, N, K); break;
      case 1: k_gemm<1><<<g, 256, 0, stream>>>(A, Bt, C_, M, N, K); break;
      case 2: k_gemm<2><<<g, 256, 0, stream>>>(A, Bt, C_, M, N, K); break;
      case 3: k_gemm<3><<<g, 256, 0, stream>>>(A, Bt, C_, M, N, K); break;
      case 4: k_gemm<4><<<g, 256, 0, stream>>>(A, Bt, C_, M, N, K); break;
    }
  };

  // --------------- weight preprocessing (f32 -> f16, transpose to [N][K]) ----
  for (int l = 0; l < Ll; ++l) {
    xpose(wq  + (size_t)l * Cc * HD, wqT + (size_t)l * HD * Cc, Cc, HD);
    xpose(wk  + (size_t)l * Cc * KD, wkT + (size_t)l * KD * Cc, Cc, KD);
    xpose(wv  + (size_t)l * Cc * KD, wvT + (size_t)l * KD * Cc, Cc, KD);
    xpose(wo  + (size_t)l * HD * Cc, woT + (size_t)l * Cc * HD, HD, Cc);
    xpose(msgw+ (size_t)l * Cc * Cc, msgT+ (size_t)l * Cc * Cc, Cc, Cc);
    xpose(lisw+ (size_t)l * Cc * Cc, lisT+ (size_t)l * Cc * Cc, Cc, Cc);
    xpose(tgw + (size_t)l * 2 * Cc * Cc,                   tgTopT + (size_t)l * Cc * Cc, Cc, Cc);
    xpose(tgw + (size_t)l * 2 * Cc * Cc + (size_t)Cc * Cc, tgBotT + (size_t)l * Cc * Cc, Cc, Cc);
    for (int e = 0; e < NEx; ++e) {
      size_t s13 = ((size_t)l * NEx + e) * Cc * Ff;
      size_t s2  = ((size_t)l * NEx + e) * Ff * Cc;
      xpose(ew1 + s13, w1T + s2,  Cc, Ff);
      xpose(ew3 + s13, w3T + s2,  Cc, Ff);
      xpose(ew2 + s2,  w2T + s13, Ff, Cc);
    }
  }
  xpose(lmh, lmT, Cc, Vv);

  // --------------- forward -------------------------------------------------
  k_gather<<<gblk((size_t)Tt * Cc), 256, 0, stream>>>(idx, wte, x);

  for (int l = 0; l < Ll; ++l) {
    // attention block
    k_ln<<<Tt, 256, 0, stream>>>(x, ln1g + l * Cc, ln1b + l * Cc, h, h16);
    gemm(0, h16, wqT + (size_t)l * HD * Cc, qb, Tt, HD, Cc);
    gemm(0, h16, wkT + (size_t)l * KD * Cc, kb, Tt, KD, Cc);
    gemm(0, h16, wvT + (size_t)l * KD * Cc, vb, Tt, KD, Cc);
    k_rope<<<gblk((size_t)Tt * Hh * 32), 256, 0, stream>>>(qb, Hh);
    k_rope<<<gblk((size_t)Tt * KVh * 32), 256, 0, stream>>>(kb, KVh);
    k_attn<<<dim3(Tt, Hh), 128, 0, stream>>>(qb, kb, vb, yb);
    k_convert<<<gblk((size_t)Tt * HD), 256, 0, stream>>>(yb, y16, (size_t)Tt * HD);
    gemm(0, y16, woT + (size_t)l * Cc * HD, ao, Tt, Cc, HD);
    k_add<<<gblk((size_t)Tt * Cc), 256, 0, stream>>>(x, ao, (size_t)Tt * Cc);

    // MoE block
    k_ln<<<Tt, 256, 0, stream>>>(x, ln2g + l * Cc, ln2b + l * Cc, h2, h2_16);
    k_gate<<<Tt, 256, 0, stream>>>(h2, agate + (size_t)l * Cc * Aa,
                                   egate + (size_t)l * Cc * NEx, expw, topw, topi);
    k_init_agent<<<gblk((size_t)Tt * Aa * Cc), 256, 0, stream>>>(agout,
                                                                 role + (size_t)l * Aa * Cc);
    for (int e = 0; e < NEx; ++e) {
      size_t sUp = ((size_t)l * NEx + e) * Ff * Cc;
      size_t sDn = ((size_t)l * NEx + e) * Cc * Ff;
      gemm(0, h2_16, w1T + sUp, ub, Tt, Ff, Cc);
      gemm(0, h2_16, w3T + sUp, gb, Tt, Ff, Cc);
      k_silu_mul<<<gblk((size_t)Tt * Ff), 256, 0, stream>>>(ub, gb, act16, (size_t)Tt * Ff);
      gemm(0, act16, w2T + sDn, yeb, Tt, Cc, Ff);
      k_acc_agent<<<gblk((size_t)Tt * Cc), 256, 0, stream>>>(yeb, expw, agout, e, e / 2);
    }

    // social block
    k_build_wout<<<gblk((size_t)T3 * Cc), 256, 0, stream>>>(agout, topw, topi, wout, wout16);
    gemm(1, wout16, msgT + (size_t)l * Cc * Cc, msgs, T3, Cc, Cc);           // tanh
    k_other<<<gblk((size_t)T3 * Cc), 256, 0, stream>>>(msgs, othr, othr16);
    gemm(0, othr16, lisT + (size_t)l * Cc * Cc, lst, T3, Cc, Cc);
    k_convert<<<gblk((size_t)T3 * Cc), 256, 0, stream>>>(lst, lst16, (size_t)T3 * Cc);
    gemm(0, wout16, tgTopT + (size_t)l * Cc * Cc, gateb, T3, Cc, Cc);
    gemm(4, lst16, tgBotT + (size_t)l * Cc * Cc, gateb, T3, Cc, Cc);         // +, sigmoid
    k_strength<<<1, 32, 0, stream>>>(topi, tstr + (size_t)l * Aa, strg);
    k_refine<<<gblk((size_t)Tt * Cc), 256, 0, stream>>>(x, wout, gateb, lst, strg);
  }

  // final LN + LM head
  k_ln<<<Tt, 256, 0, stream>>>(x, lnfg, lnfb, h, h16);
  gemm(0, h16, lmT, out, Tt, Vv, Cc);
}
